// WeatherMeshEncoder_71811853189920
// MI455X (gfx1250) — matrix operand order repaired and checked by reading, not compiled, not run on hardware
//
#include <hip/hip_runtime.h>
#include <hip/hip_bf16.h>
#include <stdint.h>

typedef __attribute__((ext_vector_type(16))) _Float16 v16h;
typedef __attribute__((ext_vector_type(8)))  float    v8f;

union Frag16 { v16h v; uint32_t u[8]; };
union Acc8   { v8f  v; float    f[8]; };
union H2     { _Float16 h[2]; uint32_t u; };

#if __has_builtin(__builtin_amdgcn_global_load_async_to_lds_b32)
#define HAVE_ASYNC 1
#else
#define HAVE_ASYNC 0
#endif

typedef __attribute__((address_space(1))) int GInt;   // global
typedef __attribute__((address_space(3))) int LInt;   // LDS

__device__ __forceinline__ void async_copy_b32(const float* g, float* l) {
#if HAVE_ASYNC
    __builtin_amdgcn_global_load_async_to_lds_b32((GInt*)g, (LInt*)l, 0, 0);
#else
    *l = *g;
#endif
}

__device__ __forceinline__ void async_wait_all() {
#if HAVE_ASYNC
#if __has_builtin(__builtin_amdgcn_s_wait_asynccnt)
    __builtin_amdgcn_s_wait_asynccnt(0);
#else
    asm volatile("s_wait_asynccnt 0x0" ::: "memory");
#endif
#endif
}

__device__ __forceinline__ uint32_t packh(float a, float b) {
    H2 p; p.h[0] = (_Float16)a; p.h[1] = (_Float16)b; return p.u;
}

__device__ __forceinline__ float gelu_tanh(float x) {
    float x3 = x * x * x;
    return 0.5f * x * (1.0f + tanhf(0.7978845608028654f * (x + 0.044715f * x3)));
}

// ---------------------------------------------------------------------------
// Stride-2 conv (3D; 2D via KD=1, PD=0, Din=1) as implicit GEMM on
// v_wmma_f32_16x16x32_f16. Block = 128 thr = 4 waves.
// Tile: M = 32 contiguous output-w pixels x N = 128 out-channels.
// Each wave: 2 A-frags x 2 B-frags -> 4 WMMAs per 32-k chunk.
// ---------------------------------------------------------------------------
template <int KD, int PD>
__global__ __launch_bounds__(128) void conv_wmma_kernel(
    const float* __restrict__ in, const float* __restrict__ wgt,
    const float* __restrict__ bias, float* __restrict__ out,
    int Cin, int Din, int Hin, int Win,
    int Cout, int Dout, int Hout, int Wout,
    int out_cstride)
{
    __shared__ uint32_t Ah32[32][16];    // 32 pixels x 32 halfs
    __shared__ uint32_t Bh32[128][16];   // 128 chans x 32 halfs

    const int tilesW = Wout >> 5;
    const int bx   = blockIdx.x;
    const int tw   = bx % tilesW;
    const int rest = bx / tilesW;
    const int oh   = rest % Hout;
    const int od   = rest / Hout;
    const int co0  = blockIdx.y * 128;
    const int ow0  = tw * 32;

    const int tid  = threadIdx.x;
    const int lane = tid & 31;
    const int wv   = tid >> 5;

    constexpr int KD9 = KD * 9;          // compile-time divisor
    const int Ktot  = Cin * KD9;
    const int nChnk = (Ktot + 31) >> 5;
    const int HWin  = Hin * Win;
    const int DHWin = Din * HWin;

    Acc8 acc[2][2];
    #pragma unroll
    for (int a = 0; a < 2; ++a)
        #pragma unroll
        for (int b = 0; b < 2; ++b)
            #pragma unroll
            for (int i = 0; i < 8; ++i) acc[a][b].f[i] = 0.0f;

    for (int kc = 0; kc < nChnk; ++kc) {
        const int k0c = kc << 5;
        // ---- stage A (im2col gather, f16): thread -> pixel tid/4, 8 consecutive k
        {
            const int m  = tid >> 2;
            const int ks = (tid & 3) << 3;
            const int ow = ow0 + m;
            float vv[8];
            #pragma unroll
            for (int j = 0; j < 8; ++j) {
                const int kk = k0c + ks + j;
                float v = 0.0f;
                if (kk < Ktot) {
                    const int ci = kk / KD9;
                    const int r  = kk - ci * KD9;
                    const int kz = r / 9;
                    const int r2 = r - kz * 9;
                    const int ky = r2 / 3;
                    const int kx = r2 - ky * 3;
                    const int id = od * 2 - PD + kz;
                    const int ih = oh * 2 - 1 + ky;
                    const int iw = ow * 2 - 1 + kx;
                    if ((unsigned)id < (unsigned)Din && (unsigned)ih < (unsigned)Hin &&
                        (unsigned)iw < (unsigned)Win)
                        v = in[(size_t)ci * DHWin + (size_t)id * HWin + ih * Win + iw];
                }
                vv[j] = v;
            }
            #pragma unroll
            for (int q = 0; q < 4; ++q)
                Ah32[m][(ks >> 1) + q] = packh(vv[2 * q], vv[2 * q + 1]);
        }
        // ---- stage B (weights, contiguous in k): thread -> one channel, 32 halfs
        {
            const float* wp = wgt + (size_t)(co0 + tid) * Ktot;
            #pragma unroll
            for (int j = 0; j < 32; j += 2) {
                const int kk = k0c + j;
                Bh32[tid][j >> 1] = packh((kk     < Ktot) ? wp[kk]     : 0.0f,
                                          (kk + 1 < Ktot) ? wp[kk + 1] : 0.0f);
            }
        }
        __syncthreads();
        // ---- fragments (ISA 16-bit A 16x32 / B 32x16 layouts) + 4 WMMAs
        {
            const int m  = lane & 15;
            const int hi = lane >> 4;
            Frag16 af[2], bf[2];
            #pragma unroll
            for (int p = 0; p < 8; ++p) {
                const int ai = (p & 3) + hi * 4 + ((p >= 4) ? 8 : 0);
                af[0].u[p] = Ah32[m][ai];
                af[1].u[p] = Ah32[m + 16][ai];
                const int bi = (p & 3) + hi * 4 + ((p >= 4) ? 8 : 0);
                bf[0].u[p] = Bh32[wv * 32 + m][bi];
                bf[1].u[p] = Bh32[wv * 32 + 16 + m][bi];
            }
            #pragma unroll
            for (int mi = 0; mi < 2; ++mi)
                #pragma unroll
                for (int ni = 0; ni < 2; ++ni)
                    acc[mi][ni].v = __builtin_amdgcn_wmma_f32_16x16x32_f16(
                        false, af[mi].v, false, bf[ni].v, (short)0,
                        acc[mi][ni].v, false, false);
        }
        __syncthreads();
    }

    // ---- epilogue: bias + GELU; C layout N=lane&15, M=r+8*(lane>=16)
    const int n  = lane & 15;
    const int hi = lane >> 4;
    #pragma unroll
    for (int ni = 0; ni < 2; ++ni) {
        const int co = co0 + wv * 32 + ni * 16 + n;
        const float bv = bias[co];
        const size_t obase =
            (size_t)co * out_cstride + ((size_t)od * Hout + oh) * Wout + ow0;
        #pragma unroll
        for (int mi = 0; mi < 2; ++mi)
            #pragma unroll
            for (int r = 0; r < 8; ++r)
                out[obase + mi * 16 + r + hi * 8] = gelu_tanh(acc[mi][ni].f[r] + bv);
    }
}

// ---------------------------------------------------------------------------
// Strided GEMM: C[m,n] = A[m,k]*B[k,n] + bias[n] (+ residual), f16 WMMA.
// Requires M%32==0, N%128==0, K%32==0 (true for all uses: K=512).
// f32 tiles staged to LDS with GLOBAL_LOAD_ASYNC_TO_LDS_B32 (ASYNCcnt),
// converted to f16 during fragment build.
// ---------------------------------------------------------------------------
__global__ __launch_bounds__(128) void gemm_wmma_kernel(
    const float* __restrict__ A, int As_m, int As_k,
    const float* __restrict__ B, int Bs_k, int Bs_n,
    const float* __restrict__ bias,
    float* __restrict__ C, int Cs_m, int Cs_n,
    const float* __restrict__ resid,
    int M, int N, int K)
{
    __shared__ float Af[32][32];     // 4 KB
    __shared__ float Bf[128][32];    // 16 KB

    const int m0 = blockIdx.x * 32;
    const int n0 = blockIdx.y * 128;
    const int tid = threadIdx.x;
    const int lane = tid & 31;
    const int wv   = tid >> 5;

    Acc8 acc[2][2];
    #pragma unroll
    for (int a = 0; a < 2; ++a)
        #pragma unroll
        for (int b = 0; b < 2; ++b)
            #pragma unroll
            for (int i = 0; i < 8; ++i) acc[a][b].f[i] = 0.0f;

    const int nChnk = K >> 5;
    for (int kc = 0; kc < nChnk; ++kc) {
        const int k0c = kc << 5;
        {   // A: thread -> row m=tid/4, 8 consecutive k (async f32 -> LDS)
            const int m  = tid >> 2;
            const int ks = (tid & 3) << 3;
            const float* ap = A + (size_t)(m0 + m) * As_m + (size_t)(k0c + ks) * As_k;
            #pragma unroll
            for (int j = 0; j < 8; ++j)
                async_copy_b32(ap + (size_t)j * As_k, &Af[m][ks + j]);
        }
        {   // B: thread -> one output column n=tid, 32 k (async f32 -> LDS)
            const float* bp = B + (size_t)(n0 + tid) * Bs_n + (size_t)k0c * Bs_k;
            #pragma unroll
            for (int j = 0; j < 32; ++j)
                async_copy_b32(bp + (size_t)j * Bs_k, &Bf[tid][j]);
        }
        async_wait_all();
        __syncthreads();
        {
            const int m  = lane & 15;
            const int hi = lane >> 4;
            Frag16 af[2], bf[2];
            #pragma unroll
            for (int p = 0; p < 8; ++p) {
                const int ai = ((p & 3) + hi * 4 + ((p >= 4) ? 8 : 0)) * 2;
                af[0].u[p] = packh(Af[m][ai],      Af[m][ai + 1]);
                af[1].u[p] = packh(Af[m + 16][ai], Af[m + 16][ai + 1]);
                const int bi = ((p & 3) + hi * 4 + ((p >= 4) ? 8 : 0)) * 2;
                bf[0].u[p] = packh(Bf[wv * 32 + m][bi],      Bf[wv * 32 + m][bi + 1]);
                bf[1].u[p] = packh(Bf[wv * 32 + 16 + m][bi], Bf[wv * 32 + 16 + m][bi + 1]);
            }
            #pragma unroll
            for (int mi = 0; mi < 2; ++mi)
                #pragma unroll
                for (int ni = 0; ni < 2; ++ni)
                    acc[mi][ni].v = __builtin_amdgcn_wmma_f32_16x16x32_f16(
                        false, af[mi].v, false, bf[ni].v, (short)0,
                        acc[mi][ni].v, false, false);
        }
        __syncthreads();
    }

    const int n  = lane & 15;
    const int hi = lane >> 4;
    #pragma unroll
    for (int ni = 0; ni < 2; ++ni) {
        const int ng = n0 + wv * 32 + ni * 16 + n;
        const float bv = bias ? bias[ng] : 0.0f;
        #pragma unroll
        for (int mi = 0; mi < 2; ++mi)
            #pragma unroll
            for (int r = 0; r < 8; ++r) {
                const int mg = m0 + mi * 16 + r + hi * 8;
                float v = acc[mi][ni].f[r] + bv;
                const size_t ci = (size_t)mg * Cs_m + (size_t)ng * Cs_n;
                if (resid) v += resid[ci];
                C[ci] = v;
            }
    }
}

// ---------------------------------------------------------------------------
// LayerNorm over C=512 per token; lat is [512][8192] (C-major), y is [8192][512].
// Block = 128 thr = 4 waves, one token per wave; shuffle reductions.
// ---------------------------------------------------------------------------
__global__ __launch_bounds__(128) void ln_kernel(
    const float* __restrict__ lat, const float* __restrict__ g,
    const float* __restrict__ b, float* __restrict__ y)
{
    const int t = blockIdx.x * 4 + (threadIdx.x >> 5);
    const int lane = threadIdx.x & 31;
    float xv[16];
    float s = 0.0f, s2 = 0.0f;
    #pragma unroll
    for (int j = 0; j < 16; ++j) {
        const float v = lat[(size_t)(lane + j * 32) * 8192 + t];
        xv[j] = v; s += v; s2 += v * v;
    }
    #pragma unroll
    for (int off = 16; off > 0; off >>= 1) {
        s  += __shfl_xor(s,  off, 32);
        s2 += __shfl_xor(s2, off, 32);
    }
    const float mu   = s * (1.0f / 512.0f);
    const float var  = s2 * (1.0f / 512.0f) - mu * mu;
    const float rstd = rsqrtf(var + 1e-5f);
    #pragma unroll
    for (int j = 0; j < 16; ++j) {
        const int c = lane + j * 32;
        y[(size_t)t * 512 + c] = (xv[j] - mu) * rstd * g[c] + b[c];
    }
}

// ---------------------------------------------------------------------------
// 3D neighborhood attention, window 5x7x7 over D=4,H=32,W=64, 8 heads, hd=64.
// qkv = [8192][1536] (q|k|v). One wave per (token, head); 2 channels/lane.
// ---------------------------------------------------------------------------
__global__ __launch_bounds__(256) void attn_kernel(
    const float* __restrict__ qkv, float* __restrict__ outp)
{
    const int t    = blockIdx.x;
    const int head = threadIdx.x >> 5;
    const int lane = threadIdx.x & 31;
    const int d = t >> 11;
    const int h = (t >> 6) & 31;
    const int w = t & 63;
    const int hoff = head * 64 + lane * 2;

    const float* qp = qkv + (size_t)t * 1536 + hoff;
    const float q0 = qp[0] * 0.125f;
    const float q1 = qp[1] * 0.125f;

    float sc[8];
    #pragma unroll
    for (int i = 0; i < 8; ++i) sc[i] = -3.0e38f;

    for (int j = 0; j < 245; ++j) {
        const int oz = j / 49;
        const int rm = j - oz * 49;
        const int oy = rm / 7;
        const int ox = rm - oy * 7;
        const int zd = d + oz - 2, yh = h + oy - 3, xw = w + ox - 3;
        const bool valid = (unsigned)zd < 4u && (unsigned)yh < 32u && (unsigned)xw < 64u;
        float dot = -3.0e38f;
        if (valid) {   // uniform across the wave
            const int t2 = (zd << 11) + (yh << 6) + xw;
            const float* kp = qkv + (size_t)t2 * 1536 + 512 + hoff;
            dot = q0 * kp[0] + q1 * kp[1];
            #pragma unroll
            for (int off = 16; off > 0; off >>= 1) dot += __shfl_xor(dot, off, 32);
        }
        if (lane == (j & 31)) sc[j >> 5] = dot;
    }

    float mx = sc[0];
    #pragma unroll
    for (int i = 1; i < 8; ++i) mx = fmaxf(mx, sc[i]);
    #pragma unroll
    for (int off = 16; off > 0; off >>= 1) mx = fmaxf(mx, __shfl_xor(mx, off, 32));
    float ssum = 0.0f;
    #pragma unroll
    for (int i = 0; i < 8; ++i) { sc[i] = expf(sc[i] - mx); ssum += sc[i]; }
    #pragma unroll
    for (int off = 16; off > 0; off >>= 1) ssum += __shfl_xor(ssum, off, 32);
    const float inv = 1.0f / ssum;

    float o0 = 0.0f, o1 = 0.0f;
    for (int j = 0; j < 245; ++j) {
        const int oz = j / 49;
        const int rm = j - oz * 49;
        const int oy = rm / 7;
        const int ox = rm - oy * 7;
        const int zd = d + oz - 2, yh = h + oy - 3, xw = w + ox - 3;
        if (!((unsigned)zd < 4u && (unsigned)yh < 32u && (unsigned)xw < 64u)) continue;
        const float pj = __shfl(sc[j >> 5], j & 31, 32) * inv;
        const int t2 = (zd << 11) + (yh << 6) + xw;
        const float* vp = qkv + (size_t)t2 * 1536 + 1024 + hoff;
        o0 += pj * vp[0];
        o1 += pj * vp[1];
    }
    float* op = outp + (size_t)t * 512 + hoff;
    op[0] = o0;
    op[1] = o1;
}

// ---------------------------------------------------------------------------
extern "C" void kernel_launch(void* const* d_in, const int* in_sizes, int n_in,
                              void* d_out, int out_size, void* d_ws, size_t ws_size,
                              hipStream_t stream) {
    const float* x2d  = (const float*)d_in[0];
    const float* x3d  = (const float*)d_in[1];
    const float* sw0  = (const float*)d_in[2];  const float* sb0 = (const float*)d_in[3];
    const float* sw1  = (const float*)d_in[4];  const float* sb1 = (const float*)d_in[5];
    const float* sw2  = (const float*)d_in[6];  const float* sb2 = (const float*)d_in[7];
    const float* pw0  = (const float*)d_in[8];  const float* pb0 = (const float*)d_in[9];
    const float* pw1  = (const float*)d_in[10]; const float* pb1 = (const float*)d_in[11];
    const float* pw2  = (const float*)d_in[12]; const float* pb2 = (const float*)d_in[13];
    const float* latw = (const float*)d_in[14]; const float* latb = (const float*)d_in[15];
    const float* lng  = (const float*)d_in[16]; const float* lnb  = (const float*)d_in[17];
    const float* qkvw = (const float*)d_in[18]; const float* qkvb = (const float*)d_in[19];
    const float* projw= (const float*)d_in[20]; const float* projb= (const float*)d_in[21];

    float* out = (float*)d_out;          // [512][8192] == [1,512,4,32,64]
    float* ws  = (float*)d_ws;

    // workspace arena (floats); p1 region reused once the 3D stack drains
    float* p1      = ws;                       // 128*12*128*256 = 50,331,648
    float* p2      = ws + 50331648;            // 256*6*64*128   = 12,582,912
    float* feats   = ws;                       // 512*4*32*64    =  4,194,304 (reuses p1)
    float* s1      = ws + 4194304;             // 128*128*256
    float* s2      = ws + 8388608;             // 256*64*128
    float* ybuf    = ws + 10485760;            // 8192*512
    float* qkvbuf  = ws + 14680064;            // 8192*1536
    float* attnout = ws + 27262976;            // 8192*512

    dim3 blk(128);
    auto conv3 = [&](const float* in, const float* wgt, const float* b, float* o,
                     int Cin, int Din, int Hin, int Win,
                     int Cout, int Dout, int Hout, int Wout, int cstr) {
        dim3 grid((Wout / 32) * Dout * Hout, Cout / 128);
        conv_wmma_kernel<3, 1><<<grid, blk, 0, stream>>>(in, wgt, b, o,
            Cin, Din, Hin, Win, Cout, Dout, Hout, Wout, cstr);
    };
    auto conv2 = [&](const float* in, const float* wgt, const float* b, float* o,
                     int Cin, int Hin, int Win,
                     int Cout, int Hout, int Wout, int cstr) {
        dim3 grid((Wout / 32) * Hout, Cout / 128);
        conv_wmma_kernel<1, 0><<<grid, blk, 0, stream>>>(in, wgt, b, o,
            Cin, 1, Hin, Win, Cout, 1, Hout, Wout, cstr);
    };

    // 3D pressure stack first (largest footprint), then reuse p1 region
    conv3(x3d, pw0, pb0, p1,    5, 24, 256, 512, 128, 12, 128, 256, 12 * 128 * 256);
    conv3(p1,  pw1, pb1, p2,  128, 12, 128, 256, 256,  6,  64, 128,  6 *  64 * 128);
    conv3(p2,  pw2, pb2, feats, 256, 6, 64, 128, 512,  3,  32,  64, 8192);   // depth 0..2
    // 2D surface stack
    conv2(x2d, sw0, sb0, s1,    8, 256, 512, 128, 128, 256, 128 * 256);
    conv2(s1,  sw1, sb1, s2,  128, 128, 256, 256,  64, 128,  64 * 128);
    conv2(s2,  sw2, sb2, feats + 3 * 2048, 256, 64, 128, 512, 32, 64, 8192); // depth 3

    // lat = 1x1x1 conv: out[co][pos] = sum_ci latw[co][ci]*feats[ci][pos] + latb[co]
    gemm_wmma_kernel<<<dim3(256, 4), blk, 0, stream>>>(
        feats, 1, 8192, latw, 1, 512, latb, out, 1, 8192, nullptr, 8192, 512, 512);

    for (int i = 0; i < 3; ++i) {
        ln_kernel<<<2048, blk, 0, stream>>>(out, lng + i * 512, lnb + i * 512, ybuf);
        gemm_wmma_kernel<<<dim3(256, 12), blk, 0, stream>>>(
            ybuf, 512, 1, qkvw + (size_t)i * 512 * 1536, 1536, 1,
            qkvb + i * 1536, qkvbuf, 1536, 1, nullptr, 8192, 1536, 512);
        attn_kernel<<<8192, 256, 0, stream>>>(qkvbuf, attnout);
        gemm_wmma_kernel<<<dim3(256, 4), blk, 0, stream>>>(
            attnout, 512, 1, projw + (size_t)i * 512 * 512, 512, 1,
            projb + i * 512, out, 1, 8192, out, 8192, 512, 512);
    }
}
